// ShiftWindowMSA_20890720928463
// MI455X (gfx1250) — matrix operand order repaired and checked by reading, not compiled, not run on hardware
//
#include <hip/hip_runtime.h>

typedef __attribute__((ext_vector_type(16))) _Float16 v16h;
typedef __attribute__((ext_vector_type(8)))  float    v8f;

#define NTOK   64
#define EMB    96
#define NH     3
#define HD     32
#define QKV_N  288
#define SHIFT_ 4

// LDS layout (bytes)
#define OFF_QKV   0                       // f16 [64][288] = 36864  (lives stages 1-4)
#define OFF_XS    36864                   // f16 [64][96]  = 12288  (stage 0-1)
#define OFF_ATTN  36864                   // f16 [64][64]  =  8192  (union w/ XS, per-head)
#define OFF_AO    49152                   // f16 [64][96]  = 12288  (stages 4-5)
#define OFF_REL   61440                   // f32 [225*3]   =  2700
#define SMEM_BYTES 64144

// ---- fragment loaders (ISA 7.12.2 layouts, wave32) ----

// A-matrix 16x32 f16, row-major source with leading dim `ld` (f16 elements).
__device__ __forceinline__ v16h load_frag_A(const _Float16* src, int ld, int row0,
                                            int k0, int lane) {
  int m  = row0 + (lane & 15);
  int kb = k0 + ((lane >> 4) << 3);          // lanes 0-15: +0, lanes 16-31: +8
  const _Float16* p = src + m * ld + kb;
  v16h a;
#pragma unroll
  for (int j = 0; j < 8; ++j) a[j] = p[j];          // K = kb .. kb+7
#pragma unroll
  for (int j = 0; j < 8; ++j) a[j + 8] = p[16 + j]; // K = kb+16 .. kb+23
  return a;
}

// B-matrix 32x16 f16: element B[k][n] at base[k*strideK + n*strideN].
// lanes 0-15: K = k0..k0+7, k0+16..k0+23; lanes 16-31: K = k0+8..k0+15, k0+24..k0+31; col = n0 + (lane&15).
__device__ __forceinline__ v16h load_frag_B(const _Float16* base, int strideK,
                                            int strideN, int n0, int k0, int lane) {
  int n  = n0 + (lane & 15);
  int kb = k0 + ((lane >> 4) << 3);
  v16h b;
#pragma unroll
  for (int j = 0; j < 16; ++j) b[j] = base[(kb + j + ((j >> 3) << 3)) * strideK + n * strideN];
  return b;
}

__device__ __forceinline__ int region9(int gr, int gc) {
  int rr = (gr < 248) ? 0 : ((gr < 252) ? 1 : 2);
  int rc = (gc < 248) ? 0 : ((gc < 252) ? 1 : 2);
  return rr * 3 + rc;
}

__global__ void swin_convert_w(const float* __restrict__ qkvw,
                               const float* __restrict__ projw,
                               _Float16* __restrict__ dst) {
  int i = blockIdx.x * blockDim.x + threadIdx.x;
  if (i < 27648)       dst[i] = (_Float16)qkvw[i];
  else if (i < 36864)  dst[i] = (_Float16)projw[i - 27648];
}

__global__ __launch_bounds__(256) void swin_msa_fused(
    const float* __restrict__ query, const float* __restrict__ qkv_b,
    const float* __restrict__ proj_b, const float* __restrict__ rel_table,
    const _Float16* __restrict__ qkvw, const _Float16* __restrict__ projw,
    float* __restrict__ out) {
  __shared__ __align__(16) char smem[SMEM_BYTES];
  _Float16* qkvs  = (_Float16*)(smem + OFF_QKV);
  _Float16* xs    = (_Float16*)(smem + OFF_XS);
  _Float16* attnH = (_Float16*)(smem + OFF_ATTN);
  _Float16* ao    = (_Float16*)(smem + OFF_AO);
  float*    relL  = (float*)(smem + OFF_REL);

  const int tid  = threadIdx.x;
  const int lane = tid & 31;
  const int wave = tid >> 5;

  const int wgid = blockIdx.x;          // 4096 = 4 imgs * 32 * 32 windows
  const int bimg = wgid >> 10;
  const int wy   = (wgid >> 5) & 31;
  const int wx   = wgid & 31;

  // ---- stage 0: stage rel table + input window (roll -shift folded in) ----
  for (int i = tid; i < 675; i += 256) relL[i] = rel_table[i];
  {
    int t  = tid >> 2;                  // token 0..63
    int c0 = (tid & 3) * 24;
    int my = t >> 3, mx = t & 7;
    int orow = (wy * 8 + my + SHIFT_) & 255;
    int ocol = (wx * 8 + mx + SHIFT_) & 255;
    const float* src = query + ((size_t)(bimg * 65536 + orow * 256 + ocol)) * EMB;
#pragma unroll
    for (int c = 0; c < 24; ++c) xs[t * EMB + c0 + c] = (_Float16)src[c0 + c];
  }
  __syncthreads();

  // ---- stage 1: QKV GEMM  [64,96] x [96,288]^T, 4x18 tiles, K=96 ----
  const float scale = 0.17677669529663687f;  // 32^-0.5
  for (int tt = wave; tt < 72; tt += 8) {
    int mt = tt & 3, nt = tt >> 2;
    v8f acc = {};
#pragma unroll
    for (int kk = 0; kk < 96; kk += 32) {
      v16h a = load_frag_A(xs, EMB, mt * 16, kk, lane);
      v16h b = load_frag_B(qkvw, 1, EMB, nt * 16, kk, lane); // B[k][o]=W[o*96+k]
      acc = __builtin_amdgcn_wmma_f32_16x16x32_f16(false, a, false, b,
                                                   (short)0, acc, false, false);
    }
    int o = nt * 16 + (lane & 15);
    float bias = qkv_b[o];
    float sc = (o < 96) ? scale : 1.0f;   // scale applies to q slice only
    int mrow = mt * 16 + ((lane >> 4) << 3);
#pragma unroll
    for (int j = 0; j < 8; ++j)
      qkvs[(mrow + j) * QKV_N + o] = (_Float16)((acc[j] + bias) * sc);
  }
  __syncthreads();

  // ---- stages 2-4, per head ----
  for (int h = 0; h < NH; ++h) {
    // QK^T: [64,32] x [32,64], 4x4 tiles, one K=32 WMMA each
    for (int tt = wave; tt < 16; tt += 8) {
      int mt = tt >> 2, nt = tt & 3;
      v16h a = load_frag_A(qkvs + h * HD, QKV_N, mt * 16, 0, lane);
      v16h b = load_frag_B(qkvs + 96 + h * HD, 1, QKV_N, nt * 16, 0, lane);
      v8f acc = {};
      acc = __builtin_amdgcn_wmma_f32_16x16x32_f16(false, a, false, b,
                                                   (short)0, acc, false, false);
      int n  = nt * 16 + (lane & 15);
      int ny = n >> 3, nx = n & 7;
      int regn = region9(wy * 8 + ny, wx * 8 + nx);
      int mrow = mt * 16 + ((lane >> 4) << 3);
#pragma unroll
      for (int j = 0; j < 8; ++j) {
        int m  = mrow + j;
        int qy = m >> 3, qx = m & 7;
        float v = acc[j] + relL[((qy - ny + 7) * 15 + (qx - nx + 7)) * NH + h];
        if (region9(wy * 8 + qy, wx * 8 + qx) != regn) v -= 100.0f;
        attnH[m * 64 + n] = (_Float16)v;
      }
    }
    __syncthreads();

    // softmax over rows (fp32 math, f16 storage, in place)
    if (tid < 64) {
      _Float16* row = attnH + tid * 64;
      float mx = -1e30f;
#pragma unroll
      for (int n = 0; n < 64; ++n) mx = fmaxf(mx, (float)row[n]);
      float s = 0.0f;
#pragma unroll
      for (int n = 0; n < 64; ++n) s += __expf((float)row[n] - mx);
      float inv = 1.0f / s;
#pragma unroll
      for (int n = 0; n < 64; ++n)
        row[n] = (_Float16)(__expf((float)row[n] - mx) * inv);
    }
    __syncthreads();

    // AV: [64,64] x [64,32], 4x2 tiles -> exactly one per wave, K=64
    {
      int mt = wave >> 1, nt = wave & 1;
      v8f acc = {};
#pragma unroll
      for (int kk = 0; kk < 64; kk += 32) {
        v16h a = load_frag_A(attnH, 64, mt * 16, kk, lane);
        v16h b = load_frag_B(qkvs + 192 + h * HD, QKV_N, 1, nt * 16, kk, lane);
        acc = __builtin_amdgcn_wmma_f32_16x16x32_f16(false, a, false, b,
                                                     (short)0, acc, false, false);
      }
      int d = nt * 16 + (lane & 15);
      int mrow = mt * 16 + ((lane >> 4) << 3);
#pragma unroll
      for (int j = 0; j < 8; ++j)
        ao[(mrow + j) * EMB + h * HD + d] = (_Float16)acc[j];
    }
    __syncthreads();
  }

  // ---- stage 5: projection [64,96] x [96,96]^T, 4x6 tiles; +shift on store ----
  for (int tt = wave; tt < 24; tt += 8) {
    int mt = tt & 3, nt = tt >> 2;
    v8f acc = {};
#pragma unroll
    for (int kk = 0; kk < 96; kk += 32) {
      v16h a = load_frag_A(ao, EMB, mt * 16, kk, lane);
      v16h b = load_frag_B(projw, 1, EMB, nt * 16, kk, lane);
      acc = __builtin_amdgcn_wmma_f32_16x16x32_f16(false, a, false, b,
                                                   (short)0, acc, false, false);
    }
    int o = nt * 16 + (lane & 15);
    float bias = proj_b[o];
    int mrow = mt * 16 + ((lane >> 4) << 3);
#pragma unroll
    for (int j = 0; j < 8; ++j) {
      int m  = mrow + j;
      int my = m >> 3, mx = m & 7;
      int orow = (wy * 8 + my + SHIFT_) & 255;
      int ocol = (wx * 8 + mx + SHIFT_) & 255;
      out[((size_t)(bimg * 65536 + orow * 256 + ocol)) * EMB + o] = acc[j] + bias;
    }
  }
}

extern "C" void kernel_launch(void* const* d_in, const int* in_sizes, int n_in,
                              void* d_out, int out_size, void* d_ws, size_t ws_size,
                              hipStream_t stream) {
  (void)in_sizes; (void)n_in; (void)out_size; (void)ws_size;
  const float* query  = (const float*)d_in[0];
  const float* qkv_w  = (const float*)d_in[1];
  const float* qkv_b  = (const float*)d_in[2];
  const float* proj_w = (const float*)d_in[3];
  const float* proj_b = (const float*)d_in[4];
  const float* rel    = (const float*)d_in[5];
  // d_in[6], d_in[7] = h, w (static 256 in this workload)

  _Float16* wsp = (_Float16*)d_ws;                 // 27648 qkv_w f16 + 9216 proj_w f16
  swin_convert_w<<<144, 256, 0, stream>>>(qkv_w, proj_w, wsp);
  swin_msa_fused<<<4096, 256, 0, stream>>>(query, qkv_b, proj_b, rel,
                                           wsp, wsp + 27648, (float*)d_out);
}
